// MAF_BiMamba_25237227832030
// MI455X (gfx1250) — matrix operand order repaired and checked by reading, not compiled, not run on hardware
//
#include <hip/hip_runtime.h>

// ---------------------------------------------------------------------------
// Types for CDNA5 WMMA (wave32, 16x16x32 bf16 -> f32 accumulate)
// ---------------------------------------------------------------------------
typedef __attribute__((ext_vector_type(16))) __bf16 v16bf;
typedef __attribute__((ext_vector_type(8)))  float  v8f;
typedef unsigned short u16;

union ABFrag { unsigned u[8]; v16bf v; };

#if defined(__gfx1250__) && __has_builtin(__builtin_amdgcn_tensor_load_to_lds)
#define USE_TDM 1
#else
#define USE_TDM 0
#endif

__device__ inline u16 f2bf(float x) {
    unsigned u = __float_as_uint(x);
    u = (u + 0x7fffu + ((u >> 16) & 1u)) >> 16;   // RNE
    return (u16)u;
}
__device__ inline float gelu_f(float x) {
    return 0.5f * x * (1.f + erff(x * 0.7071067811865476f));
}
__device__ inline float silu_f(float x) {
    return x / (1.f + __expf(-x));
}
__device__ inline int imin(int a, int b){ return a < b ? a : b; }
__device__ inline int imax(int a, int b){ return a > b ? a : b; }

#if USE_TDM
typedef __attribute__((ext_vector_type(4))) unsigned tdm_u4;
typedef __attribute__((ext_vector_type(8))) int      tdm_i8;
typedef __attribute__((ext_vector_type(4))) int      tdm_i4;

// 2D bf16 tile load: 32 elements (64B) x 128 rows, global row stride ld_elems,
// LDS pitch padded 64B+16B = 80B (pad_interval=16 dwords, pad_amount=4 dwords).
// rows_rem < 128 rows are zero-filled via tensor_dim OOB handling.
__device__ inline void tdm_load_2d(unsigned lds_off, const u16* gptr,
                                   int rows_rem, int ld_elems)
{
    unsigned long long ga = (unsigned long long)(size_t)gptr;
    tdm_u4 g0;
    g0[0] = 1u;                                            // count=1 (valid), user mode
    g0[1] = lds_off;                                       // lds_addr
    g0[2] = (unsigned)(ga & 0xffffffffull);                // global_addr[31:0]
    g0[3] = (unsigned)((ga >> 32) & 0x01ffffffull)         // global_addr[56:32]
          | 0x80000000u;                                   // type=2 (image)
    tdm_i8 g1;
    g1[0] = (1 << 16)        // data_size = 2 bytes
          | (1 << 20)        // pad_enable
          | (3 << 22)        // pad_interval: 16 dwords (64B)
          | (3 << 25);       // pad_amount: 4 dwords (16B)
    g1[1] = (32 & 0xffff) << 16;                 // tensor_dim0[15:0] = 32
    g1[2] = ((rows_rem & 0xffff) << 16);         // tensor_dim0 hi=0 | tensor_dim1 lo
    g1[3] = ((rows_rem >> 16) & 0xffff)          // tensor_dim1 hi
          | (32 << 16);                          // tile_dim0 = 32 elements
    g1[4] = 128;                                 // tile_dim1 = 128, tile_dim2 = 0
    g1[5] = ld_elems;                            // tensor_dim0_stride[31:0]
    g1[6] = 0;
    g1[7] = 0;
    tdm_i4 z4; z4[0] = 0; z4[1] = 0; z4[2] = 0; z4[3] = 0;
    tdm_i8 z8;
    #pragma unroll
    for (int i = 0; i < 8; ++i) z8[i] = 0;
    __builtin_amdgcn_tensor_load_to_lds(g0, g1, z4, z4, z8, 0);
}
#endif

// ---------------------------------------------------------------------------
// WMMA GEMM:  C[M,N] (+)= A[M,K] @ W[N,K]^T (+ bias);  A,W bf16; C f32 and/or bf16
// Block tile 128x128, 8 waves (2x4), each wave 4(M)x2(N) 16x16 tiles, K step 32.
// Tiles staged into double-buffered LDS by the Tensor Data Mover (wave 0 issues,
// s_wait_tensorcnt + barriers synchronize; TDM of step s+1 overlaps WMMA of s).
// ---------------------------------------------------------------------------
#define BM 128
#define BN 128
#define LDP 20   // LDS row pitch in dwords (80B) -> conflict-free frag reads

__global__ __launch_bounds__(256) void gemm_bf16_wmma(
    const u16* __restrict__ A, int lda,
    const u16* __restrict__ W, int ldw,
    float* __restrict__ C, int ldc,
    const float* __restrict__ bias,
    u16* __restrict__ Cbf,
    int M, int N, int K, int accum)
{
    __shared__ unsigned sA[2][BM * LDP];
    __shared__ unsigned sB[2][BN * LDP];

    const int tid  = threadIdx.x;
    const int m0   = blockIdx.y * BM;
    const int n0   = blockIdx.x * BN;
    const int wave = tid >> 5;
    const int lane = tid & 31;
    const int wm   = wave >> 2;   // 0..1 -> 64 rows each
    const int wn   = wave & 3;    // 0..3 -> 32 cols each
    const int lr   = lane & 15;
    const int lh   = lane >> 4;

    int aRem = M - m0; if (aRem > BM) aRem = BM;
    int bRem = N - n0; if (bRem > BN) bRem = BN;

    v8f c[4][2];
    #pragma unroll
    for (int i = 0; i < 4; ++i)
        #pragma unroll
        for (int j = 0; j < 2; ++j)
            #pragma unroll
            for (int r = 0; r < 8; ++r) c[i][j][r] = 0.f;

    const int nsteps = K >> 5;

#if USE_TDM
    if (tid < 32) {   // wave 0 issues TDM for first K-step (EXEC!=0 for whole wave)
        tdm_load_2d((unsigned)(unsigned long long)(size_t)&sA[0][0],
                    A + (size_t)m0 * lda, aRem, lda);
        tdm_load_2d((unsigned)(unsigned long long)(size_t)&sB[0][0],
                    W + (size_t)n0 * ldw, bRem, ldw);
    }
#endif

    for (int s = 0; s < nsteps; ++s) {
        const int cur = s & 1;
#if USE_TDM
        if (tid < 32) {
            if (s + 1 < nsteps) {     // prefetch next K-step into other buffer
                int kn = (s + 1) << 5;
                tdm_load_2d((unsigned)(unsigned long long)(size_t)&sA[cur ^ 1][0],
                            A + (size_t)m0 * lda + kn, aRem, lda);
                tdm_load_2d((unsigned)(unsigned long long)(size_t)&sB[cur ^ 1][0],
                            W + (size_t)n0 * ldw + kn, bRem, ldw);
                __builtin_amdgcn_s_wait_tensorcnt((short)2);  // current pair done (in-order)
            } else {
                __builtin_amdgcn_s_wait_tensorcnt((short)0);
            }
        }
#else
        {   // fallback: manual bf16 dword copy into LDS
            int k0 = s << 5;
            #pragma unroll
            for (int it = 0; it < 8; ++it) {
                int p   = tid + it * 256;   // 128 rows x 16 dwords
                int row = p >> 4;
                int cp  = p & 15;
                unsigned av = 0u, bv = 0u;
                if (row < aRem) av = *(const unsigned*)(A + (size_t)(m0 + row) * lda + k0 + cp * 2);
                if (row < bRem) bv = *(const unsigned*)(W + (size_t)(n0 + row) * ldw + k0 + cp * 2);
                sA[cur][row * LDP + cp] = av;
                sB[cur][row * LDP + cp] = bv;
            }
        }
#endif
        __syncthreads();

        // B fragments (32x16: lane half h holds K = 16h..16h+15 contiguous)
        ABFrag bfr[2];
        #pragma unroll
        for (int j = 0; j < 2; ++j) {
            int nr = wn * 32 + j * 16 + lr;
            const unsigned* bp = &sB[cur][nr * LDP + 8 * lh];
            #pragma unroll
            for (int q = 0; q < 8; ++q) bfr[j].u[q] = bp[q];
        }
        // A fragments (16x32: lane half h holds K = 16h..16h+15 contiguous)
        #pragma unroll
        for (int i = 0; i < 4; ++i) {
            int rr = wm * 64 + i * 16 + lr;
            ABFrag af;
            const unsigned* ap = &sA[cur][rr * LDP + 8 * lh];
            #pragma unroll
            for (int q = 0; q < 4; ++q) { af.u[q] = ap[q]; af.u[q + 4] = ap[q + 4]; }
            #pragma unroll
            for (int j = 0; j < 2; ++j) {
                c[i][j] = __builtin_amdgcn_wmma_f32_16x16x32_bf16(
                    false, af.v, false, bfr[j].v, (short)0, c[i][j], false, false);
            }
        }
        __syncthreads();   // all waves done reading before this buffer is re-filled
    }

    // Epilogue: C/D 16x16 layout -> VGPR r holds M = r + 8*lh, N = lane%16.
    #pragma unroll
    for (int i = 0; i < 4; ++i) {
        #pragma unroll
        for (int j = 0; j < 2; ++j) {
            int n = n0 + wn * 32 + j * 16 + lr;
            if (n >= N) continue;
            float bv = bias ? bias[n] : 0.f;
            #pragma unroll
            for (int r = 0; r < 8; ++r) {
                int m = m0 + wm * 64 + i * 16 + r + 8 * lh;
                if (m < M) {
                    size_t idx = (size_t)m * ldc + n;
                    float v = c[i][j][r] + bv;
                    if (accum) { v += C[idx]; C[idx] = v; }
                    else if (C) C[idx] = v;
                    if (Cbf) Cbf[idx] = f2bf(v);
                }
            }
        }
    }
}

// ---------------------------------------------------------------------------
// f32 -> bf16 elementwise convert (used for weights)
// ---------------------------------------------------------------------------
__global__ void f32_to_bf16_kernel(const float* __restrict__ in, u16* __restrict__ out, int n)
{
    int i = blockIdx.x * blockDim.x + threadIdx.x;
    if (i < n) out[i] = f2bf(in[i]);
}

// ---------------------------------------------------------------------------
// NCHW -> (B*HW, C) permute, bf16 output
// ---------------------------------------------------------------------------
__global__ void nchw_to_mhc_bf(const float* __restrict__ in, u16* __restrict__ out,
                               int C, int HW, int total)
{
    int idx = blockIdx.x * blockDim.x + threadIdx.x;
    if (idx >= total) return;
    int c = idx % C;
    int m = idx / C;          // m = b*HW + p
    int b = m / HW;
    int p = m % HW;
    out[(size_t)m * C + c] = f2bf(in[((size_t)b * C + c) * HW + p]);
}

// ---------------------------------------------------------------------------
// Bilinear upsample 14x14 -> 28x28 (align_corners=False), bf16 out into cols
// 256..511 of fused [12544, 512] bf16 buffer.
// ---------------------------------------------------------------------------
__global__ void bilinear_up_kernel(const float* __restrict__ src,  // [3136,256] f32
                                   u16* __restrict__ dst)          // [12544,512] bf16
{
    int idx = blockIdx.x * blockDim.x + threadIdx.x;
    if (idx >= 16 * 784 * 256) return;
    int c  = idx & 255;
    int p  = idx >> 8;
    int b  = p / 784;
    int pp = p % 784;
    int oh = pp / 28, ow = pp % 28;
    float fh = (oh + 0.5f) * 0.5f - 0.5f;
    float fw = (ow + 0.5f) * 0.5f - 0.5f;
    int h0i = (int)floorf(fh); float ah = fh - (float)h0i;
    int w0i = (int)floorf(fw); float aw = fw - (float)w0i;
    int h0 = imin(imax(h0i, 0), 13), h1 = imin(imax(h0i + 1, 0), 13);
    int w0 = imin(imax(w0i, 0), 13), w1 = imin(imax(w0i + 1, 0), 13);
    const float* sb = src + (size_t)b * 196 * 256;
    float v00 = sb[(h0 * 14 + w0) * 256 + c];
    float v01 = sb[(h0 * 14 + w1) * 256 + c];
    float v10 = sb[(h1 * 14 + w0) * 256 + c];
    float v11 = sb[(h1 * 14 + w1) * 256 + c];
    float v = (1.f - ah) * ((1.f - aw) * v00 + aw * v01)
            +        ah  * ((1.f - aw) * v10 + aw * v11);
    dst[(size_t)p * 512 + 256 + c] = f2bf(v);
}

// ---------------------------------------------------------------------------
// LayerNorm over 512-wide rows (f32 -> f32)
// ---------------------------------------------------------------------------
__global__ __launch_bounds__(256) void ln512_kernel(
    const float* __restrict__ src, float* __restrict__ dst,
    const float* __restrict__ g, const float* __restrict__ b, int M)
{
    __shared__ float red[256];
    int m = blockIdx.x;
    if (m >= M) return;
    const float* row = src + (size_t)m * 512;
    int t = threadIdx.x;
    float v0 = row[t], v1 = row[t + 256];
    red[t] = v0 + v1;
    __syncthreads();
    for (int s = 128; s > 0; s >>= 1) { if (t < s) red[t] += red[t + s]; __syncthreads(); }
    float mean = red[0] * (1.f / 512.f);
    __syncthreads();
    float d0 = v0 - mean, d1 = v1 - mean;
    red[t] = d0 * d0 + d1 * d1;
    __syncthreads();
    for (int s = 128; s > 0; s >>= 1) { if (t < s) red[t] += red[t + s]; __syncthreads(); }
    float rstd = rsqrtf(red[0] * (1.f / 512.f) + 1e-5f);
    float* o = dst + (size_t)m * 512;
    o[t]       = d0 * rstd * g[t]       + b[t];
    o[t + 256] = d1 * rstd * g[t + 256] + b[t + 256];
}

// ---------------------------------------------------------------------------
// Mean over L=784 of [16,784,512] -> [16,512]
// ---------------------------------------------------------------------------
__global__ void mean_l_kernel(const float* __restrict__ x, float* __restrict__ out)
{
    int b = blockIdx.x;
    int c = blockIdx.y * 256 + threadIdx.x;
    float s = 0.f;
    for (int l = 0; l < 784; ++l) s += x[((size_t)(b * 784 + l)) * 512 + c];
    out[b * 512 + c] = s * (1.f / 784.f);
}

__device__ inline void meanvar_serial(const float* a, int n, float* st) {
    float m = 0.f;
    for (int i = 0; i < n; ++i) m += a[i];
    m /= (float)n;
    float v = 0.f;
    for (int i = 0; i < n; ++i) { float d = a[i] - m; v += d * d; }
    v /= (float)n;
    st[0] = m;
    st[1] = rsqrtf(v + 1e-5f);
}

// ---------------------------------------------------------------------------
// Meta encoder: one block per sample
// ---------------------------------------------------------------------------
__global__ __launch_bounds__(256) void meta_enc_kernel(
    const float* __restrict__ mn, const int* __restrict__ mc,
    const float* __restrict__ emb0, const float* __restrict__ emb1, const float* __restrict__ emb2,
    const float* __restrict__ np_g1, const float* __restrict__ np_b1,
    const float* __restrict__ np_W,  const float* __restrict__ np_b,
    const float* __restrict__ np_g2, const float* __restrict__ np_beta2,
    const float* __restrict__ fm_g1, const float* __restrict__ fm_b1,
    const float* __restrict__ fm_W1, const float* __restrict__ fm_bb1,
    const float* __restrict__ fm_g2, const float* __restrict__ fm_b2,
    const float* __restrict__ fm_W2, const float* __restrict__ fm_bb2,
    const float* __restrict__ fm_g3, const float* __restrict__ fm_b3,
    float* __restrict__ mv)
{
    __shared__ float sa[160], sb[160], st[2];
    int b = blockIdx.x, t = threadIdx.x;

    if (t < 8) sa[t] = mn[b * 8 + t];
    __syncthreads();
    if (t == 0) meanvar_serial(sa, 8, st);
    __syncthreads();
    if (t < 8) sb[t] = (sa[t] - st[0]) * st[1] * np_g1[t] + np_b1[t];
    __syncthreads();
    if (t < 64) {
        float s = np_b[t];
        for (int k = 0; k < 8; ++k) s += np_W[t * 8 + k] * sb[k];
        sa[t] = gelu_f(s);
    }
    __syncthreads();
    if (t == 0) meanvar_serial(sa, 64, st);
    __syncthreads();
    if (t < 64) sb[t] = (sa[t] - st[0]) * st[1] * np_g2[t] + np_beta2[t];
    if (t < 96) {
        int e = t / 32, j = t % 32;
        int dim = (e == 0 ? 10 : (e == 1 ? 5 : 8));
        int col = imin(imax(mc[b * 3 + e], 0), dim - 1);
        const float* E = (e == 0 ? emb0 : (e == 1 ? emb1 : emb2));
        sb[64 + t] = E[col * 32 + j];
    }
    __syncthreads();
    if (t == 0) meanvar_serial(sb, 160, st);
    __syncthreads();
    if (t < 160) sa[t] = (sb[t] - st[0]) * st[1] * fm_g1[t] + fm_b1[t];
    __syncthreads();
    if (t < 128) {
        float s = fm_bb1[t];
        for (int k = 0; k < 160; ++k) s += fm_W1[t * 160 + k] * sa[k];
        sb[t] = gelu_f(s);
    }
    __syncthreads();
    if (t == 0) meanvar_serial(sb, 128, st);
    __syncthreads();
    if (t < 128) sa[t] = (sb[t] - st[0]) * st[1] * fm_g2[t] + fm_b2[t];
    __syncthreads();
    if (t < 64) {
        float s = fm_bb2[t];
        for (int k = 0; k < 128; ++k) s += fm_W2[t * 128 + k] * sa[k];
        sb[t] = gelu_f(s);
    }
    __syncthreads();
    if (t == 0) meanvar_serial(sb, 64, st);
    __syncthreads();
    if (t < 64) mv[b * 64 + t] = (sb[t] - st[0]) * st[1] * fm_g3[t] + fm_b3[t];
}

// ---------------------------------------------------------------------------
// FiLM MLP: one block per sample. Produces g (0..511) and beta (512..1023).
// ---------------------------------------------------------------------------
__global__ __launch_bounds__(256) void film_mlp_kernel(
    const float* __restrict__ xmean, const float* __restrict__ mv,
    const float* __restrict__ mp_W, const float* __restrict__ mp_b,
    const float* __restrict__ mp_g, const float* __restrict__ mp_beta,
    const float* __restrict__ ip_W, const float* __restrict__ ip_b,
    const float* __restrict__ ip_g, const float* __restrict__ ip_beta,
    const float* __restrict__ fg1_W, const float* __restrict__ fg1_b,
    const float* __restrict__ fg2_W, const float* __restrict__ fg2_b,
    float* __restrict__ gb)
{
    __shared__ float sx[512], smv[64], su[256], sv[256], st[4];
    int b = blockIdx.x, t = threadIdx.x;
    sx[t] = xmean[b * 512 + t];
    sx[t + 256] = xmean[b * 512 + 256 + t];
    if (t < 64) smv[t] = mv[b * 64 + t];
    __syncthreads();
    if (t < 128) {                 // ie -> su[128..255]
        float s = ip_b[t];
        for (int k = 0; k < 512; ++k) s += ip_W[t * 512 + k] * sx[k];
        su[128 + t] = gelu_f(s);
    } else {                       // me -> su[0..127]
        int tt = t - 128;
        float s = mp_b[tt];
        for (int k = 0; k < 64; ++k) s += mp_W[tt * 64 + k] * smv[k];
        su[tt] = gelu_f(s);
    }
    __syncthreads();
    if (t == 0) meanvar_serial(su, 128, st);
    if (t == 1) meanvar_serial(su + 128, 128, st + 2);
    __syncthreads();
    float nv;
    if (t < 128) nv = (su[t] - st[0]) * st[1] * mp_g[t] + mp_beta[t];
    else { int tt = t - 128; nv = (su[t] - st[2]) * st[3] * ip_g[tt] + ip_beta[tt]; }
    __syncthreads();
    su[t] = nv;
    __syncthreads();
    {
        float s = fg1_b[t];
        for (int k = 0; k < 256; ++k) s += fg1_W[t * 256 + k] * su[k];
        sv[t] = gelu_f(s);
    }
    __syncthreads();
    #pragma unroll
    for (int q = 0; q < 4; ++q) {
        int o = q * 256 + t;
        float s = fg2_b[o];
        for (int k = 0; k < 256; ++k) s += fg2_W[o * 256 + k] * sv[k];
        gb[b * 1024 + o] = s;
    }
}

// ---------------------------------------------------------------------------
// FiLM apply: xm_bf = bf16( xn*(1+g[b]) + beta[b] )   (bf16-only output; the
// FiLM result feeds only the in_proj GEMM)
// ---------------------------------------------------------------------------
__global__ void film_apply_kernel(const float* __restrict__ xn,
                                  const float* __restrict__ gb,
                                  u16* __restrict__ out_bf)
{
    int idx = blockIdx.x * blockDim.x + threadIdx.x;
    if (idx >= 12544 * 512) return;
    int c = idx & 511;
    int m = idx >> 9;
    int b = m / 784;
    float gv = gb[b * 1024 + c];
    float bv = gb[b * 1024 + 512 + c];
    out_bf[idx] = f2bf(xn[idx] * (1.f + gv) + bv);
}

// ---------------------------------------------------------------------------
// Causal depthwise conv(k=4) + SiLU; writes f32 (for scan) and bf16 (for GEMM).
// dir=1 emulates flip+conv+flip by indexing l..l+3 with reversed taps.
// ---------------------------------------------------------------------------
__global__ void conv_silu_kernel(const float* __restrict__ xz,
                                 const float* __restrict__ cw,
                                 const float* __restrict__ cb,
                                 float* __restrict__ out,
                                 u16* __restrict__ out_bf, int dir)
{
    int idx = blockIdx.x * blockDim.x + threadIdx.x;
    if (idx >= 12544 * 1024) return;
    int c = idx & 1023;
    int m = idx >> 10;
    int l = m % 784;
    int b = m / 784;
    float s = cb[c];
    #pragma unroll
    for (int j = 0; j < 4; ++j) {
        int li = dir ? (l + 3 - j) : (l - 3 + j);
        if (li >= 0 && li < 784)
            s += cw[c * 4 + j] * xz[((size_t)(b * 784 + li)) * 2048 + c];
    }
    float v = silu_f(s);
    out[(size_t)m * 1024 + c]    = v;
    out_bf[(size_t)m * 1024 + c] = f2bf(v);
}

// ---------------------------------------------------------------------------
// Selective scan: softplus + SSM recurrence + skip (D) + SiLU(z) gate fused.
// grid (4,16); dir=1 runs reversed sequence writing y at original positions.
// Output bf16 (consumed only by the out_proj GEMM).
// ---------------------------------------------------------------------------
__global__ __launch_bounds__(256) void scan_kernel(
    const float* __restrict__ dpre,   // [12544,1024] delta pre-softplus
    const float* __restrict__ dbc,    // [12544,64]   cols 32..47 = B, 48..63 = C
    const float* __restrict__ xcs,    // [12544,1024]
    const float* __restrict__ xz,     // [12544,2048] z in cols 1024..2047
    const float* __restrict__ A_log,  // [1024,16]
    const float* __restrict__ Dv,     // [1024]
    u16* __restrict__ y, int dir)
{
    int b = blockIdx.y;
    int d = blockIdx.x * 256 + threadIdx.x;
    float Arow[16], h[16];
    #pragma unroll
    for (int s = 0; s < 16; ++s) {
        Arow[s] = -__expf(A_log[d * 16 + s]);
        h[s] = 0.f;
    }
    float Dd = Dv[d];
    __shared__ float sBC[32];
    for (int t = 0; t < 784; ++t) {
        int l = dir ? (783 - t) : t;
        size_t m = (size_t)b * 784 + l;
        __syncthreads();
        if (threadIdx.x < 32) sBC[threadIdx.x] = dbc[m * 64 + 32 + threadIdx.x];
        __syncthreads();
        float dp = dpre[m * 1024 + d];
        float delta = dp > 20.f ? dp : log1pf(__expf(dp));
        float xv = xcs[m * 1024 + d];
        float dx = delta * xv;
        float acc = 0.f;
        #pragma unroll
        for (int s = 0; s < 16; ++s) {
            h[s] = __expf(delta * Arow[s]) * h[s] + dx * sBC[s];
            acc += h[s] * sBC[16 + s];
        }
        float zv = xz[m * 2048 + 1024 + d];
        y[m * 1024 + d] = f2bf((acc + xv * Dd) * silu_f(zv));
    }
}

// ---------------------------------------------------------------------------
// Head: LN(mean feature) + FC -> [16, 8]
// ---------------------------------------------------------------------------
__global__ __launch_bounds__(256) void head_kernel(
    const float* __restrict__ xmean,
    const float* __restrict__ hg, const float* __restrict__ hb,
    const float* __restrict__ fcW, const float* __restrict__ fcb,
    float* __restrict__ out)
{
    __shared__ float sfeat[512], red[256];
    int b = blockIdx.x, t = threadIdx.x;
    float v0 = xmean[b * 512 + t], v1 = xmean[b * 512 + 256 + t];
    red[t] = v0 + v1;
    __syncthreads();
    for (int s = 128; s > 0; s >>= 1) { if (t < s) red[t] += red[t + s]; __syncthreads(); }
    float mean = red[0] * (1.f / 512.f);
    __syncthreads();
    float d0 = v0 - mean, d1 = v1 - mean;
    red[t] = d0 * d0 + d1 * d1;
    __syncthreads();
    for (int s = 128; s > 0; s >>= 1) { if (t < s) red[t] += red[t + s]; __syncthreads(); }
    float rstd = rsqrtf(red[0] * (1.f / 512.f) + 1e-5f);
    __syncthreads();
    sfeat[t]       = d0 * rstd * hg[t]       + hb[t];
    sfeat[t + 256] = d1 * rstd * hg[t + 256] + hb[t + 256];
    __syncthreads();
    int o = t >> 5, lane = t & 31;
    float p = 0.f;
    for (int c = lane; c < 512; c += 32) p += sfeat[c] * fcW[o * 512 + c];
    red[t] = p;
    __syncthreads();
    if (lane == 0) {
        float s = 0.f;
        for (int i = 0; i < 32; ++i) s += red[o * 32 + i];
        out[b * 8 + o] = s + fcb[o];
    }
}

// ---------------------------------------------------------------------------
// Host-side launch
// ---------------------------------------------------------------------------
extern "C" void kernel_launch(void* const* d_in, const int* in_sizes, int n_in,
                              void* d_out, int out_size, void* d_ws, size_t ws_size,
                              hipStream_t stream)
{
    (void)in_sizes; (void)n_in; (void)out_size; (void)ws_size;
    const int ML = 16 * 784;   // 12544 rows

    auto F = [&](int i) { return (const float*)d_in[i]; };
    const float* f_fine   = F(0);
    const float* f_coarse = F(1);
    const float* meta_num = F(2);
    const int*   meta_cat = (const int*)d_in[3];

    // ---- workspace layout: f32 region then bf16 region ----
    float* ws = (float*)d_ws;
    size_t o = 0;
    float* x     = ws + o; o += (size_t)ML * 512;
    float* xn    = ws + o; o += (size_t)ML * 512;
    float* xz    = ws + o; o += (size_t)ML * 2048;
    float* xcs   = ws + o; o += (size_t)ML * 1024;
    float* dbc   = ws + o; o += (size_t)ML * 64;
    float* delta = ws + o; o += (size_t)ML * 1024;
    float* accb  = ws + o; o += (size_t)ML * 512;
    float* mv    = ws + o; o += 16 * 64;
    float* xmean = ws + o; o += 16 * 512;
    float* gb    = ws + o; o += 16 * 1024;

    u16* bf = (u16*)(ws + o);
    size_t ob = 0;
    u16* xn_bf   = bf + ob; ob += (size_t)ML * 512;
    u16* xcs_bf  = bf + ob; ob += (size_t)ML * 1024;
    u16* dbc_bf  = bf + ob; ob += (size_t)ML * 64;
    u16* y_bf    = bf + ob; ob += (size_t)ML * 1024;
    u16* acc_bf  = bf + ob; ob += (size_t)ML * 512;
    u16* fu2d_bf = bf + ob; ob += (size_t)ML * 512;
    u16* wbf     = bf + ob; ob += (size_t)2048 * 512;   // largest weight

    // fusion-stage aliases (stream-serialized, safe)
    u16*   Af_bf = xcs_bf;   // [12544, 256] bf16
    u16*   Ac_bf = y_bf;     // [3136, 512] bf16
    float* fcp   = delta;    // [3136, 256] f32

    auto gemm = [&](const u16* A, int lda, const u16* W, int ldw,
                    float* C, int ldc, const float* bias, u16* Cbf,
                    int M, int N, int K, int accum) {
        dim3 g((N + 127) / 128, (M + 127) / 128);
        gemm_bf16_wmma<<<g, 256, 0, stream>>>(A, lda, W, ldw, C, ldc, bias, Cbf,
                                              M, N, K, accum);
    };
    auto convW = [&](int i, int elems) {
        f32_to_bf16_kernel<<<(elems + 255) / 256, 256, 0, stream>>>(F(i), wbf, elems);
    };

    // ---- Fusion front-end ----
    {
        int tot_f = 16 * 256 * 784;
        nchw_to_mhc_bf<<<(tot_f + 255) / 256, 256, 0, stream>>>(f_fine, Af_bf, 256, 784, tot_f);
        convW(4, 256 * 256);
        gemm(Af_bf, 256, wbf, 256, nullptr, 512, F(5), fu2d_bf, ML, 256, 256, 0);
        int tot_c = 16 * 512 * 196;
        nchw_to_mhc_bf<<<(tot_c + 255) / 256, 256, 0, stream>>>(f_coarse, Ac_bf, 512, 196, tot_c);
        convW(6, 256 * 512);
        gemm(Ac_bf, 512, wbf, 512, fcp, 256, F(7), nullptr, 3136, 256, 512, 0);
        bilinear_up_kernel<<<(16 * 784 * 256 + 255) / 256, 256, 0, stream>>>(fcp, fu2d_bf);
        convW(8, 512 * 512);
        gemm(fu2d_bf, 512, wbf, 512, x, 512, F(9), nullptr, ML, 512, 512, 0);
    }

    // ---- Meta encoder ----
    meta_enc_kernel<<<16, 256, 0, stream>>>(
        meta_num, meta_cat,
        F(10), F(11), F(12),
        F(13), F(14), F(15), F(16), F(17), F(18),
        F(19), F(20), F(21), F(22), F(23), F(24), F(25), F(26), F(27), F(28),
        mv);

    // ---- BiMamba blocks ----
    for (int L = 0; L < 6; ++L) {
        int pb = 29 + 34 * L;
        ln512_kernel<<<ML, 256, 0, stream>>>(x, xn, F(pb + 0), F(pb + 1), ML);
        mean_l_kernel<<<dim3(16, 2), 256, 0, stream>>>(xn, xmean);
        film_mlp_kernel<<<16, 256, 0, stream>>>(
            xmean, mv,
            F(pb + 2), F(pb + 3), F(pb + 4), F(pb + 5),
            F(pb + 6), F(pb + 7), F(pb + 8), F(pb + 9),
            F(pb + 10), F(pb + 11), F(pb + 12), F(pb + 13), gb);
        film_apply_kernel<<<(ML * 512 + 255) / 256, 256, 0, stream>>>(xn, gb, xn_bf);

        for (int dir = 0; dir < 2; ++dir) {
            int mb = pb + (dir ? 23 : 14);
            convW(mb + 0, 2048 * 512);
            gemm(xn_bf, 512, wbf, 512, xz, 2048, nullptr, nullptr, ML, 2048, 512, 0);
            conv_silu_kernel<<<(ML * 1024 + 255) / 256, 256, 0, stream>>>(
                xz, F(mb + 1), F(mb + 2), xcs, xcs_bf, dir);
            convW(mb + 3, 64 * 1024);
            gemm(xcs_bf, 1024, wbf, 1024, dbc, 64, nullptr, dbc_bf, ML, 64, 1024, 0);
            convW(mb + 4, 1024 * 32);
            gemm(dbc_bf, 64, wbf, 32, delta, 1024, F(mb + 5), nullptr, ML, 1024, 32, 0);
            scan_kernel<<<dim3(4, 16), 256, 0, stream>>>(
                delta, dbc, xcs, xz, F(mb + 6), F(mb + 7), y_bf, dir);
            convW(mb + 8, 512 * 1024);
            gemm(y_bf, 1024, wbf, 1024, accb, 512, nullptr,
                 dir ? acc_bf : nullptr, ML, 512, 1024, dir);   // bwd accumulates, emits bf16
        }
        convW(pb + 32, 512 * 512);
        gemm(acc_bf, 512, wbf, 512, x, 512, F(pb + 33), nullptr, ML, 512, 512, 1);
    }

    // ---- Head ----
    mean_l_kernel<<<dim3(16, 2), 256, 0, stream>>>(x, xmean);
    head_kernel<<<16, 256, 0, stream>>>(xmean, F(233), F(234), F(235), F(236), (float*)d_out);
}
